// DeformableConv1d_6433861009933
// MI455X (gfx1250) — matrix operand order repaired and checked by reading, not compiled, not run on hardware
//
#include <hip/hip_runtime.h>

// ---- problem constants (match reference) ----
#define B_    8
#define C_    128
#define L_    16384
#define KT    3
#define KRED  (C_ * KT)     // 384
#define NKC   (KRED / 32)   // 12 chunks of 32 along reduction dim (order: tap*128 + c)
#define EPS_  1e-5f

typedef __attribute__((ext_vector_type(8)))  float   v8f;
typedef __attribute__((ext_vector_type(4)))  float   f32x4;
typedef __attribute__((ext_vector_type(16))) __bf16  v16bf;

static __device__ __forceinline__ unsigned short f2bf(float f) {
  union { float f; unsigned u; } v; v.f = f;
  unsigned r = v.u + 0x7FFFu + ((v.u >> 16) & 1u);   // round-to-nearest-even
  return (unsigned short)(r >> 16);
}

static __device__ __forceinline__ float wave_sum(float v) {
  #pragma unroll
  for (int m = 16; m > 0; m >>= 1) v += __shfl_xor(v, m, 32);
  return v;
}

// -------------------------------------------------------------------------
// Kernel 0: x [B,C,L] -> xT [B,L,C]  (LDS tiled transpose, 32x32)
// -------------------------------------------------------------------------
__global__ void k_transpose(const float* __restrict__ x, float* __restrict__ xT) {
  __shared__ float s[32][33];
  const int b  = blockIdx.z;
  const int c0 = blockIdx.y * 32;
  const int l0 = blockIdx.x * 32;
  const int tx = threadIdx.x, ty = threadIdx.y;
  const float* xp = x  + (size_t)b * C_ * L_;
  float*       xo = xT + (size_t)b * L_ * C_;
  #pragma unroll
  for (int i = ty; i < 32; i += 8)
    s[i][tx] = xp[(size_t)(c0 + i) * L_ + (l0 + tx)];
  __syncthreads();
  #pragma unroll
  for (int i = ty; i < 32; i += 8)
    xo[(size_t)(l0 + i) * C_ + (c0 + tx)] = s[tx][i];
}

// -------------------------------------------------------------------------
// Kernel W: dc_w [O=128][C=128][K=3] f32 -> bf16, pre-swizzled into the
// WMMA 16-bit A-fragment (16x32) lane layout per (m-tile, k-chunk):
//   Wswz[((mt*NKC + kc)*32 + lane)*16 + j]
//   lane: m = lane%16 ; half j: K = (j<8?0:16) + (lane/16)*8 + (j%8)
//   reduction index kr = kc*32 + K  maps to (tap = kr/128, c = kr%128)
// -------------------------------------------------------------------------
__global__ void k_wprep(const float* __restrict__ dc_w, unsigned short* __restrict__ Wswz) {
  const int idx  = blockIdx.x * 256 + threadIdx.x;      // < 8*12*32*16 = 49152
  const int j    = idx & 15;
  const int lane = (idx >> 4) & 31;
  const int cidx = idx >> 9;                            // 0..95
  const int kc   = cidx % NKC;
  const int mt   = cidx / NKC;
  const int o    = mt * 16 + (lane & 15);
  const int Kidx = ((j < 8) ? 0 : 16) + (lane >> 4) * 8 + (j & 7);
  const int kr   = kc * 32 + Kidx;                      // 0..383
  const int tap  = kr >> 7;                             // /128
  const int c    = kr & 127;
  Wswz[idx] = f2bf(dc_w[(o * C_ + c) * KT + tap]);
}

// -------------------------------------------------------------------------
// Kernel 1: depthwise conv + LayerNorm + ReLU + offset linear.
// One wave32 per (b,l); 4 channels per lane; wave shuffles for reductions.
// Writes fractional sample positions pos[b][l][k] = l + (k-1) + offset_k.
// -------------------------------------------------------------------------
__global__ void k_offsets(const float* __restrict__ xT,
                          const float* __restrict__ dw_w, const float* __restrict__ dw_b,
                          const float* __restrict__ ln_g, const float* __restrict__ ln_b,
                          const float* __restrict__ off_w, const float* __restrict__ off_b,
                          float* __restrict__ pos) {
  const int wave = threadIdx.x >> 5;
  const int lane = threadIdx.x & 31;
  const int g    = blockIdx.x * 8 + wave;
  const int b    = g / L_;
  const int l    = g % L_;
  const int c    = lane * 4;

  const float* base = xT + (size_t)b * L_ * C_;
  f32x4 xm = (l > 0)      ? *(const f32x4*)(base + (size_t)(l - 1) * C_ + c) : (f32x4)0.0f;
  f32x4 x0 =                *(const f32x4*)(base + (size_t)l       * C_ + c);
  f32x4 xp = (l < L_ - 1) ? *(const f32x4*)(base + (size_t)(l + 1) * C_ + c) : (f32x4)0.0f;

  float y[4];
  #pragma unroll
  for (int q = 0; q < 4; ++q) {
    const int cq = c + q;
    y[q] = dw_w[cq * KT + 0] * xm[q] + dw_w[cq * KT + 1] * x0[q]
         + dw_w[cq * KT + 2] * xp[q] + dw_b[cq];
  }
  // LayerNorm over 128 channels
  float mu = wave_sum(y[0] + y[1] + y[2] + y[3]) * (1.0f / C_);
  float vs = 0.f;
  #pragma unroll
  for (int q = 0; q < 4; ++q) { float d = y[q] - mu; vs += d * d; }
  const float rstd = rsqrtf(wave_sum(vs) * (1.0f / C_) + EPS_);
  float yr[4];
  #pragma unroll
  for (int q = 0; q < 4; ++q) {
    float yn = (y[q] - mu) * rstd * ln_g[c + q] + ln_b[c + q];
    yr[q] = fmaxf(yn, 0.0f);
  }
  // offset linear C -> 3
  #pragma unroll
  for (int k = 0; k < KT; ++k) {
    float ok = 0.f;
    #pragma unroll
    for (int q = 0; q < 4; ++q) ok += yr[q] * off_w[k * C_ + c + q];
    ok = wave_sum(ok);
    if (lane == 0)
      pos[((size_t)b * L_ + l) * KT + k] = (float)l + (float)(k - 1) + ok + off_b[k];
  }
}

// -------------------------------------------------------------------------
// Kernel 2: deformable GEMM with bf16 WMMA.
//   D[o][l] = sum_{kr=(tap,c)} W[o][kr] * interp(xT[b][pos(l,tap)][c])
// 512 threads (16 waves); block tile 128(o) x 128(l); wave (mt=w&7, nh=w>>3)
// owns 16x64 -> acc[4] (32 VGPRs). Double-buffered LDS B-fragments; gather of
// chunk kc+1 overlaps WMMA of kc; one barrier per chunk. Interp params are
// wave-uniform per row: scalarized with readfirstlane so the gather issues in
// saddr (GVS) form.
// -------------------------------------------------------------------------
__global__ void k_deform(const float* __restrict__ xT, const float* __restrict__ pos,
                         const unsigned short* __restrict__ Wswz, float* __restrict__ out) {
  __shared__ __align__(32) unsigned short Bfrag[2][8 * 32 * 16];  // 2 x 8 KB
  __shared__ int4 pk[KT][128];   // {byte_off0, byte_off1, w0_bits, w1_bits}

  const int tid  = threadIdx.x;
  const int wave = tid >> 5;
  const int lane = tid & 31;
  const int mt   = wave & 7;        // M-tile (output-channel group of 16)
  const int nh   = wave >> 3;       // N-half (4 of the 8 N-subtiles)
  const int b    = blockIdx.x / (L_ / 128);
  const int l0   = (blockIdx.x % (L_ / 128)) * 128;
  const char* xb = (const char*)(xT + (size_t)b * L_ * C_);

  // ---- interpolation params for all 3 taps, once; byte offsets premultiplied ----
  if (tid < KT * 128) {
    const int tap = tid >> 7;       // 0..2
    const int ll  = tid & 127;
    const float p  = pos[((size_t)b * L_ + l0 + ll) * KT + tap];
    const float p0 = floorf(p);
    const float fr = p - p0;
    const int   i0 = (int)p0;
    const int   i1 = i0 + 1;
    const float v0 = (i0 >= 0 && i0 < L_) ? 1.0f : 0.0f;
    const float v1 = (i1 >= 0 && i1 < L_) ? 1.0f : 0.0f;
    pk[tap][ll] = make_int4(min(max(i0, 0), L_ - 1) * (C_ * 4),
                            min(max(i1, 0), L_ - 1) * (C_ * 4),
                            __float_as_int((1.0f - fr) * v0),
                            __float_as_int(fr * v1));
  }
  __syncthreads();

  v8f acc[4];
  #pragma unroll
  for (int i = 0; i < 4; ++i) acc[i] = (v8f)0.0f;

  // ---- builder: interpolated B tile (K=32 x N=128) in WMMA B-fragment layout.
  // wave handles 8 rows; lane == channel-in-chunk (K index); params scalarized.
  auto build = [&](int kc, int buf) {
    const int tap = kc >> 2;
    const int c0  = (kc & 3) * 32;
    #pragma unroll
    for (int r = 0; r < 8; ++r) {
      const int ll = wave * 8 + r;                 // local position 0..127
      const int4 p = pk[tap][ll];                  // wave-uniform LDS read
      const int   o0 = __builtin_amdgcn_readfirstlane(p.x);
      const int   o1 = __builtin_amdgcn_readfirstlane(p.y);
      const float w0 = __int_as_float(__builtin_amdgcn_readfirstlane(p.z));
      const float w1 = __int_as_float(__builtin_amdgcn_readfirstlane(p.w));
      const float x0v = *(const float*)(xb + o0 + (c0 + lane) * 4);  // saddr + v-offset
      const float x1v = *(const float*)(xb + o1 + (c0 + lane) * 4);
      const float v   = w0 * x0v + w1 * x1v;
      const int n  = ll & 15;
      const int nt = ll >> 4;
      const int bl = (((lane >> 3) & 1) << 4) + n; // fragment lane = ((K/8)%2)*16 + N
      const int bj = ((lane >> 4) << 3) + (lane & 7); // half index = (K/16)*8 + K%8
      Bfrag[buf][(nt * 32 + bl) * 16 + bj] = f2bf(v);
    }
  };

  build(0, 0);
  __syncthreads();

  for (int kc = 0; kc < NKC; ++kc) {
    const int buf = kc & 1;
    // gather next chunk into the other buffer while this chunk runs WMMA
    if (kc + 1 < NKC) build(kc + 1, buf ^ 1);

    const v16bf afrag =
        *(const v16bf*)(Wswz + ((size_t)(mt * NKC + kc) * 32 + lane) * 16);
    if (kc + 1 < NKC)
      __builtin_prefetch(Wswz + ((size_t)(mt * NKC + kc + 1) * 32 + lane) * 16, 0, 3);

    #pragma unroll
    for (int i = 0; i < 4; ++i) {
      const int nt = nh * 4 + i;
      const v16bf bfrag = *(const v16bf*)(&Bfrag[buf][(nt * 32 + lane) * 16]);
      acc[i] = __builtin_amdgcn_wmma_f32_16x16x32_bf16(
          /*neg_a=*/false, afrag, /*neg_b=*/false, bfrag,
          /*c_mod=*/(short)0, acc[i], /*reuse_a=*/false, /*reuse_b=*/false);
    }
    __syncthreads();   // buf consumed by all; (buf^1) fully built for next iter
  }

  // ---- store D per ISA f32 C/D layout: VGPR r -> row r + 8*(lane/16) ----
  const int obase = mt * 16 + ((lane >> 4) << 3);
  #pragma unroll
  for (int i = 0; i < 4; ++i) {
    const int lcol = l0 + (nh * 4 + i) * 16 + (lane & 15);
    #pragma unroll
    for (int r = 0; r < 8; ++r) {
      out[((size_t)(b * C_ + obase + r)) * L_ + lcol] = acc[i][r];
    }
  }
}

// -------------------------------------------------------------------------
extern "C" void kernel_launch(void* const* d_in, const int* in_sizes, int n_in,
                              void* d_out, int out_size, void* d_ws, size_t ws_size,
                              hipStream_t stream) {
  (void)in_sizes; (void)n_in; (void)out_size; (void)ws_size;
  const float* x     = (const float*)d_in[0];
  const float* dw_w  = (const float*)d_in[1];
  const float* dw_b  = (const float*)d_in[2];
  const float* ln_g  = (const float*)d_in[3];
  const float* ln_b  = (const float*)d_in[4];
  const float* off_w = (const float*)d_in[5];
  const float* off_b = (const float*)d_in[6];
  const float* dc_w  = (const float*)d_in[7];
  float* out = (float*)d_out;

  char* ws = (char*)d_ws;
  const size_t xT_bytes  = (size_t)B_ * L_ * C_ * sizeof(float);   // 64 MB
  const size_t pos_bytes = (size_t)B_ * L_ * KT * sizeof(float);   // 1.5 MB
  float*          xT   = (float*)ws;
  float*          pos  = (float*)(ws + xT_bytes);
  unsigned short* Wswz = (unsigned short*)(ws + xT_bytes + pos_bytes); // 96 KB, 32B aligned

  k_transpose<<<dim3(L_ / 32, C_ / 32, B_), dim3(32, 8), 0, stream>>>(x, xT);
  k_wprep   <<<dim3((8 * NKC * 32 * 16) / 256), dim3(256), 0, stream>>>(dc_w, Wswz);
  k_offsets <<<dim3(B_ * L_ / 8), dim3(256), 0, stream>>>(xT, dw_w, dw_b, ln_g, ln_b,
                                                          off_w, off_b, pos);
  k_deform  <<<dim3(B_ * (L_ / 128)), dim3(512), 0, stream>>>(xT, pos, Wswz, out);
}
